// LeakyParallel_40346922778936
// MI455X (gfx1250) — matrix operand order repaired and checked by reading, not compiled, not run on hardware
//
#include <hip/hip_runtime.h>
#include <hip/hip_bf16.h>

typedef __attribute__((ext_vector_type(16))) __bf16 v16bf;
typedef __attribute__((ext_vector_type(8)))  __bf16 v8bf;
typedef __attribute__((ext_vector_type(8)))  float  v8f;

#define L_DIM 256
#define N_DIM 64
#define HIN   784
#define KP    800   // zero-padded K for w_ih (global bf16 buffer), 25 * 32
#define LDA_S 808   // LDS stride for staged A tile (bank-conflict pad)
#define H_DIM 512
#define LDH   520   // LDS stride for hidden state (bank-conflict pad)

// ---------------- kernel 0: fp32 -> bf16 weight conversion ----------------
__global__ void k_convert(const float* __restrict__ w_ih,
                          const float* __restrict__ w_hh,
                          __bf16* __restrict__ wih_b,   // [512][800], zero-padded
                          __bf16* __restrict__ whh_b) { // [512][512]
  int idx = blockIdx.x * blockDim.x + threadIdx.x;
  const int n_wih = H_DIM * KP;
  const int n_whh = H_DIM * H_DIM;
  if (idx < n_wih) {
    int r = idx / KP, k = idx - r * KP;
    float v = (k < HIN) ? w_ih[(size_t)r * HIN + k] : 0.0f;
    wih_b[idx] = (__bf16)v;
  } else if (idx < n_wih + n_whh) {
    int j = idx - n_wih;
    whh_b[j] = (__bf16)w_hh[j];
  }
}

// ---------------- kernel 1: x_proj = input @ w_ih^T + b_ih + b_hh ----------------
// grid: 512 blocks (32-row M tile each = 2 WMMA M-tiles), 256 threads = 8 waves.
// Each wave: 2 M-tiles x 4 N-tiles -> every B fragment feeds 2 WMMAs.
__global__ void __launch_bounds__(256)
k_xproj(const float* __restrict__ x,       // [16384, 784] (L*N rows)
        const __bf16* __restrict__ wih_b,  // [512, 800] bf16, zero-padded
        const float* __restrict__ b_ih,
        const float* __restrict__ b_hh,
        float* __restrict__ xproj) {       // [16384, 512] = d_out staging
  __shared__ __bf16 As[32 * LDA_S];
  const int tid = threadIdx.x;
  const int m0  = blockIdx.x * 32;

  // stage + convert the 32x784 A tile to bf16 (zero pad K to LDA_S)
  for (int idx = tid; idx < 32 * LDA_S; idx += 256) {
    int r = idx / LDA_S, k = idx - r * LDA_S;
    float v = (k < HIN) ? __builtin_nontemporal_load(&x[(size_t)(m0 + r) * HIN + k])
                        : 0.0f;
    As[idx] = (__bf16)v;
  }
  __syncthreads();

  const int lane = tid & 31;
  const int wave = tid >> 5;   // 0..7
  const int col  = lane & 15;  // A row (M) / B col (N) / C col (N)
  const int g    = lane >> 4;  // half-wave group

  v8f acc[2][4];
  #pragma unroll
  for (int i = 0; i < 4; ++i) {
    int n = wave * 64 + i * 16 + col;
    float bias = b_ih[n] + b_hh[n];   // per-column bias, constant over the 8 C rows
    #pragma unroll
    for (int j = 0; j < 8; ++j) { acc[0][i][j] = bias; acc[1][i][j] = bias; }
  }

  #pragma unroll
  for (int kk = 0; kk < 25; ++kk) {
    const int k0 = kk * 32;
    v16bf a[2];
    #pragma unroll
    for (int mt = 0; mt < 2; ++mt) {
      // A frag: lane holds M=col, K = k0+16g..+15 (contiguous, matches B)
      const int r = (mt * 16 + col) * LDA_S;
      v8bf alo = *reinterpret_cast<const v8bf*>(&As[r + k0 + 16 * g]);
      v8bf ahi = *reinterpret_cast<const v8bf*>(&As[r + k0 + 16 * g + 8]);
      a[mt] = __builtin_shufflevector(alo, ahi, 0,1,2,3,4,5,6,7,8,9,10,11,12,13,14,15);
    }
    #pragma unroll
    for (int i = 0; i < 4; ++i) {
      int n = wave * 64 + i * 16 + col;
      // B frag: lane holds N=col, K = k0 + 16g .. +15 (contiguous 32B)
      v16bf b = *reinterpret_cast<const v16bf*>(&wih_b[(size_t)n * KP + k0 + 16 * g]);
      acc[0][i] = __builtin_amdgcn_wmma_f32_16x16x32_bf16(
          false, a[0], false, b, (short)0, acc[0][i], false, false);
      acc[1][i] = __builtin_amdgcn_wmma_f32_16x16x32_bf16(
          false, a[1], false, b, (short)0, acc[1][i], false, false);
    }
  }

  // C layout: vgpr j, lane l: row = 8g + j, col = l%16
  #pragma unroll
  for (int mt = 0; mt < 2; ++mt) {
    #pragma unroll
    for (int i = 0; i < 4; ++i) {
      int n = wave * 64 + i * 16 + col;
      #pragma unroll
      for (int j = 0; j < 8; ++j) {
        int row = m0 + mt * 16 + 8 * g + j;
        xproj[(size_t)row * H_DIM + n] = acc[mt][i][j];
      }
    }
  }
}

// ---------------- kernel 2: recurrent scan, 4 independent batch chains ----------------
// grid: 4 blocks x 256 threads; block b owns batch rows [16b, 16b+16).
// Hidden state double-buffered in LDS as bf16; w_hh bf16 streamed from L2.
__global__ void __launch_bounds__(256)
k_recur(float* __restrict__ io,            // [256*64, 512]: in x_proj, out spikes
        const __bf16* __restrict__ whh_b,  // [512, 512] bf16
        const float* __restrict__ thr_p,
        const float* __restrict__ grad_p) {
  extern __shared__ __bf16 Hs[];  // [2][16][LDH]
  const int tid = threadIdx.x;
  const int blk = blockIdx.x;     // 0..3
  const float thr  = thr_p[0];
  const float grad = grad_p[0];

  for (int idx = tid; idx < 2 * 16 * LDH; idx += 256) Hs[idx] = (__bf16)0.0f;
  __syncthreads();

  const int lane = tid & 31;
  const int wave = tid >> 5;   // 0..7 -> 4 N-tiles each (64 cols)
  const int col  = lane & 15;
  const int g    = lane >> 4;

  for (int t = 0; t < L_DIM; ++t) {
    const __bf16* hr = &Hs[(t & 1) * 16 * LDH];
    __bf16*       hw = &Hs[((t + 1) & 1) * 16 * LDH];

    v8f acc[4];
    #pragma unroll
    for (int i = 0; i < 4; ++i)
      #pragma unroll
      for (int j = 0; j < 8; ++j) acc[i][j] = 0.0f;

    #pragma unroll
    for (int kk = 0; kk < 16; ++kk) {
      const int k0 = kk * 32;
      v8bf alo = *reinterpret_cast<const v8bf*>(&hr[col * LDH + k0 + 16 * g]);
      v8bf ahi = *reinterpret_cast<const v8bf*>(&hr[col * LDH + k0 + 16 * g + 8]);
      v16bf a = __builtin_shufflevector(alo, ahi, 0,1,2,3,4,5,6,7,8,9,10,11,12,13,14,15);
      #pragma unroll
      for (int i = 0; i < 4; ++i) {
        int n = wave * 64 + i * 16 + col;
        v16bf b = *reinterpret_cast<const v16bf*>(&whh_b[(size_t)n * H_DIM + k0 + 16 * g]);
        acc[i] = __builtin_amdgcn_wmma_f32_16x16x32_bf16(
            false, a, false, b, (short)0, acc[i], false, false);
      }
    }

    // epilogue: mem = recur + x_proj; h = relu(mem); spike = (h > thr) ? grad : 0
    #pragma unroll
    for (int i = 0; i < 4; ++i) {
      int n = wave * 64 + i * 16 + col;
      #pragma unroll
      for (int j = 0; j < 8; ++j) {
        int lm = 8 * g + j;                              // local batch row
        size_t row = (size_t)t * N_DIM + blk * 16 + lm;  // global row in io
        float mem = acc[i][j] + io[row * H_DIM + n];     // x_proj: L2-resident, RT load
        float h = mem > 0.0f ? mem : 0.0f;
        __builtin_nontemporal_store((h > thr) ? grad : 0.0f, &io[row * H_DIM + n]);
        hw[lm * LDH + n] = (__bf16)h;
      }
    }
    __syncthreads();  // h_t visible to all waves of this chain before t+1
  }
}

extern "C" void kernel_launch(void* const* d_in, const int* in_sizes, int n_in,
                              void* d_out, int out_size, void* d_ws, size_t ws_size,
                              hipStream_t stream) {
  (void)in_sizes; (void)n_in; (void)out_size; (void)ws_size;
  const float* input_ = (const float*)d_in[0];
  const float* w_ih   = (const float*)d_in[1];
  const float* b_ih   = (const float*)d_in[2];
  const float* w_hh   = (const float*)d_in[3];
  const float* b_hh   = (const float*)d_in[4];
  const float* thr    = (const float*)d_in[5];
  const float* grad   = (const float*)d_in[6];

  __bf16* wih_b = (__bf16*)d_ws;                                        // 512*800*2 B
  __bf16* whh_b = (__bf16*)((char*)d_ws + (size_t)H_DIM * KP * 2);      // 512*512*2 B

  const int conv_n = H_DIM * KP + H_DIM * H_DIM;
  k_convert<<<(conv_n + 255) / 256, 256, 0, stream>>>(w_ih, w_hh, wih_b, whh_b);
  k_xproj<<<512, 256, 0, stream>>>(input_, wih_b, b_ih, b_hh, (float*)d_out);
  k_recur<<<4, 256, 2 * 16 * LDH * (int)sizeof(__bf16), stream>>>(
      (float*)d_out, whh_b, thr, grad);
}
